// MambaBlock_58059367907936
// MI455X (gfx1250) — compile-verified, matrix operand order checked by reading, not run on hardware
//
#include <hip/hip_runtime.h>
#include <hip/hip_bf16.h>

// ---------------------------------------------------------------------------
// Mamba block forward for MI455X (gfx1250).
// bf16 WMMA GEMMs, double-buffered LDS fed by GLOBAL_LOAD_ASYNC_TO_LDS_B128
// (ASYNCcnt path), fast-math fused epilogues, unrolled gated-EMA scan,
// fused LayerNorm.
// ---------------------------------------------------------------------------

typedef __bf16  v16bf __attribute__((ext_vector_type(16)));
typedef float   v8f   __attribute__((ext_vector_type(8)));

static constexpr int BATCH   = 8;
static constexpr int SEQ     = 2048;
static constexpr int D_MODEL = 1024;
static constexpr int D_INNER = 2048;
static constexpr int D_CONV  = 4;
static constexpr int M_ROWS  = BATCH * SEQ;          // 16384

// GEMM tiling: 128x128x32 block, 8 waves (4 M x 2 N), wave = 32x64 = 2x4 WMMA
static constexpr int BM = 128, BN = 128, BK = 32;
static constexpr int LDT = BK + 8;   // LDS row stride in bf16 (80B: 16B aligned, conflict-free)

union Frag16 { v16bf v; uint4 u[2]; };
union Pack4  { __bf16 h[4]; uint2 u; };

// Fast transcendentals: v_exp_f32 / v_rcp_f32 / v_log_f32 hardware rates,
// ~1 ulp — well inside bf16-GEMM noise, ~5x fewer VALU ops than libm forms.
__device__ __forceinline__ float sigmoid_fast(float x) {
  return __builtin_amdgcn_rcpf(1.0f + __expf(-x));
}
__device__ __forceinline__ float silu_fast(float x) { return x * sigmoid_fast(x); }
__device__ __forceinline__ float softplus_fast(float x) {
  return (x > 20.0f) ? x : __logf(1.0f + __expf(x));
}

// Async-DMA one 32-byte row chunk (two b128) global -> LDS. INST_OFFSET is
// added to BOTH the LDS and global address, so one address pair serves both.
__device__ __forceinline__ void async_stage32(const __bf16* g, __bf16* l) {
  const unsigned la = (unsigned)(size_t)l;   // LDS flat addr: low 32 bits = LDS offset
  asm volatile("global_load_async_to_lds_b128 %0, %1, off\n\t"
               "global_load_async_to_lds_b128 %0, %1, off offset:16"
               :: "v"(la), "v"(g) : "memory");
}
__device__ __forceinline__ void wait_async0() {
  asm volatile("s_wait_asynccnt 0x0" ::: "memory");
}

// ---------------------------------------------------------------------------
// f32 -> bf16 bulk convert (for x)
// ---------------------------------------------------------------------------
__global__ __launch_bounds__(256) void cvt_bf16_kernel(
    const float* __restrict__ src, __bf16* __restrict__ dst) {
  const size_t i = ((size_t)blockIdx.x * blockDim.x + threadIdx.x) * 4;
  const float4 f = *(const float4*)(src + i);
  Pack4 pk;
  pk.h[0] = (__bf16)f.x; pk.h[1] = (__bf16)f.y;
  pk.h[2] = (__bf16)f.z; pk.h[3] = (__bf16)f.w;
  *(uint2*)(dst + i) = pk.u;
}

// ---------------------------------------------------------------------------
// Weight transpose + f32->bf16: src (R x C f32, row-major) -> dst (C x R bf16)
// ---------------------------------------------------------------------------
__global__ __launch_bounds__(256) void transpose_cvt_kernel(
    const float* __restrict__ src, __bf16* __restrict__ dst, int R, int C) {
  __shared__ float tile[32][33];
  const int c0 = blockIdx.x * 32, r0 = blockIdx.y * 32;
  const int x = threadIdx.x, y = threadIdx.y;
  #pragma unroll
  for (int i = 0; i < 32; i += 8)
    tile[y + i][x] = src[(size_t)(r0 + y + i) * C + (c0 + x)];
  __syncthreads();
  #pragma unroll
  for (int i = 0; i < 32; i += 8)
    dst[(size_t)(c0 + y + i) * R + (r0 + x)] = (__bf16)tile[x][y + i];
}

// ---------------------------------------------------------------------------
// Tiled WMMA GEMM:  C(MxN) = A(MxK,bf16) * Bt(NxK,bf16)^T  (+ fused epilogue)
//   MODE 0: in-proj  -> out0 = silu(v) [n<D_INNER] (u, f32), out1 = silu(v) (g)
//   MODE 1: dt-proj  -> out0 = sigmoid(softplus(v))   (alpha, f32)
//   MODE 2: out-proj -> out0 = v + resid              (h, f32)
// Double-buffered LDS; tiles DMA'd via async-to-LDS while WMMAs run.
// ---------------------------------------------------------------------------
template <int MODE>
__global__ __launch_bounds__(256) void gemm_wmma_kernel(
    const __bf16* __restrict__ A, const __bf16* __restrict__ Bt,
    const float* __restrict__ bias,
    float* __restrict__ out0, float* __restrict__ out1,
    const float* __restrict__ resid,
    int M, int N, int K) {

  __shared__ __bf16 As[2][BM * LDT];
  __shared__ __bf16 Bs[2][BN * LDT];

  const int tid   = threadIdx.x;
  const int wave  = tid >> 5;
  const int lane  = tid & 31;
  const int wm    = wave >> 1;       // 0..3  (M, 32 rows each)
  const int wn    = wave & 1;        // 0..1  (N, 64 cols each)
  const int mlane = lane & 15;
  const int khalf = lane >> 4;

  const int row0 = blockIdx.y * BM;
  const int col0 = blockIdx.x * BN;

  // staging map: each thread owns one 32B chunk: row = tid>>1, col half = tid&1
  const int r_st  = tid >> 1;
  const int hc_st = (tid & 1) * 16;
  const __bf16* gA = A  + (size_t)(row0 + r_st) * K + hc_st;
  const __bf16* gB = Bt + (size_t)(col0 + r_st) * K + hc_st;
  const int l_st = r_st * LDT + hc_st;

  v8f acc[2][4];
  #pragma unroll
  for (int i = 0; i < 2; ++i)
    #pragma unroll
    for (int j = 0; j < 4; ++j)
      #pragma unroll
      for (int r = 0; r < 8; ++r) acc[i][j][r] = 0.0f;

  // prologue: DMA tile 0 into buffer 0
  async_stage32(gA, &As[0][l_st]);
  async_stage32(gB, &Bs[0][l_st]);

  int buf = 0;
  for (int k0 = 0; k0 < K; k0 += BK, buf ^= 1) {
    wait_async0();        // my async writes into As/Bs[buf] have landed
    __syncthreads();      // everyone's have

    // kick off next tile's DMA into the other buffer (clamped on last iter:
    // harmless re-fetch of the final tile into the dead buffer, no branch)
    int kn = k0 + BK;
    if (kn >= K) kn = k0;
    async_stage32(gA + kn, &As[buf ^ 1][l_st]);
    async_stage32(gB + kn, &Bs[buf ^ 1][l_st]);

    Frag16 af[2], bf[4];
    #pragma unroll
    for (int i = 0; i < 2; ++i) {
      const int base = (wm * 32 + i * 16 + mlane) * LDT + khalf * 8;
      af[i].u[0] = *(const uint4*)(&As[buf][base]);
      af[i].u[1] = *(const uint4*)(&As[buf][base + 16]);
    }
    #pragma unroll
    for (int j = 0; j < 4; ++j) {
      const int base = (wn * 64 + j * 16 + mlane) * LDT + khalf * 8;
      bf[j].u[0] = *(const uint4*)(&Bs[buf][base]);
      bf[j].u[1] = *(const uint4*)(&Bs[buf][base + 16]);
    }
    #pragma unroll
    for (int i = 0; i < 2; ++i)
      #pragma unroll
      for (int j = 0; j < 4; ++j)
        acc[i][j] = __builtin_amdgcn_wmma_f32_16x16x32_bf16(
            false, af[i].v, false, bf[j].v, (short)0, acc[i][j], false, false);
  }

  // ---- fused epilogue (fast-math activations) ----
  #pragma unroll
  for (int i = 0; i < 2; ++i) {
    #pragma unroll
    for (int j = 0; j < 4; ++j) {
      const int gcol = col0 + wn * 64 + j * 16 + mlane;
      const float b  = bias[gcol];
      #pragma unroll
      for (int r = 0; r < 8; ++r) {
        const int grow = row0 + wm * 32 + i * 16 + khalf * 8 + r;
        float v = acc[i][j][r] + b;
        if constexpr (MODE == 0) {
          const float s = silu_fast(v);
          if (gcol < D_INNER) out0[(size_t)grow * D_INNER + gcol]             = s;
          else                out1[(size_t)grow * D_INNER + (gcol - D_INNER)] = s;
        } else if constexpr (MODE == 1) {
          out0[(size_t)grow * N + gcol] = sigmoid_fast(softplus_fast(v));  // alpha
        } else {
          out0[(size_t)grow * N + gcol] = v + resid[(size_t)grow * N + gcol];
        }
      }
    }
  }
}

// ---------------------------------------------------------------------------
// Depthwise causal conv1d (K=4) + bias + SiLU. u f32 in, xs bf16 out.
// ---------------------------------------------------------------------------
__global__ __launch_bounds__(256) void conv_silu_kernel(
    const float* __restrict__ u, const float* __restrict__ cw,
    const float* __restrict__ cb, __bf16* __restrict__ xs) {
  const size_t idx = (size_t)blockIdx.x * blockDim.x + threadIdx.x;
  const int d = (int)(idx % D_INNER);
  const size_t rem = idx / D_INNER;
  const int t = (int)(rem % SEQ);
  const int b = (int)(rem / SEQ);
  float acc = cb[d];
  #pragma unroll
  for (int k = 0; k < D_CONV; ++k) {
    const int tt = t - (D_CONV - 1) + k;
    if (tt >= 0)
      acc += cw[d * D_CONV + k] * u[((size_t)b * SEQ + tt) * D_INNER + d];
  }
  xs[idx] = (__bf16)silu_fast(acc);
}

// ---------------------------------------------------------------------------
// Gated EMA scan along time + skip + gate -> yy (bf16). Lane per (b,channel).
// Unrolled by 4: 12 independent loads per group are clause-issued ahead of the
// 4 dependent recurrence steps, hiding memory latency under the serial chain.
// ---------------------------------------------------------------------------
__global__ __launch_bounds__(256) void scan_kernel(
    const __bf16* __restrict__ xs, const float* __restrict__ alpha,
    const float* __restrict__ g, const float* __restrict__ Dskip,
    __bf16* __restrict__ yy) {
  const int gid = blockIdx.x * blockDim.x + threadIdx.x;   // 0 .. B*D_INNER-1
  const int d = gid % D_INNER;
  const int b = gid / D_INNER;
  const float dsk = Dskip[d];
  const size_t base0 = (size_t)b * SEQ * D_INNER + d;

  float y = 0.0f;
  for (int t = 0; t < SEQ; t += 4) {
    float xv[4], av[4], gv[4];
    #pragma unroll
    for (int q = 0; q < 4; ++q) {
      const size_t off = base0 + (size_t)(t + q) * D_INNER;
      xv[q] = (float)xs[off];
      av[q] = alpha[off];
      gv[q] = g[off];
    }
    __bf16 ov[4];
    #pragma unroll
    for (int q = 0; q < 4; ++q) {
      y = (t + q == 0) ? xv[q] : fmaf(av[q], xv[q], (1.0f - av[q]) * y);
      ov[q] = (__bf16)((y + xv[q] * dsk) * gv[q]);
    }
    #pragma unroll
    for (int q = 0; q < 4; ++q)
      yy[base0 + (size_t)(t + q) * D_INNER] = ov[q];
  }
}

// ---------------------------------------------------------------------------
// LayerNorm over D_MODEL=1024; one 256-thread block per row (wave32 shfl).
// ---------------------------------------------------------------------------
__global__ __launch_bounds__(256) void layernorm_kernel(
    const float* __restrict__ h, const float* __restrict__ ln_g,
    const float* __restrict__ ln_b, float* __restrict__ out) {
  const int row = blockIdx.x;
  const int tid = threadIdx.x;
  const size_t base = (size_t)row * D_MODEL + tid * 4;

  const float4 v = *(const float4*)(h + base);
  float s1 = v.x + v.y + v.z + v.w;
  float s2 = v.x * v.x + v.y * v.y + v.z * v.z + v.w * v.w;
  #pragma unroll
  for (int o = 16; o > 0; o >>= 1) {
    s1 += __shfl_xor(s1, o, 32);
    s2 += __shfl_xor(s2, o, 32);
  }
  __shared__ float r1[8], r2[8];
  const int wv = tid >> 5, lane = tid & 31;
  if (lane == 0) { r1[wv] = s1; r2[wv] = s2; }
  __syncthreads();
  float t1 = 0.0f, t2 = 0.0f;
  #pragma unroll
  for (int w = 0; w < 8; ++w) { t1 += r1[w]; t2 += r2[w]; }
  const float mu   = t1 * (1.0f / D_MODEL);
  const float var  = t2 * (1.0f / D_MODEL) - mu * mu;
  const float rstd = rsqrtf(var + 1e-5f);

  const int c = tid * 4;
  const float4 gg = *(const float4*)(ln_g + c);
  const float4 bb = *(const float4*)(ln_b + c);
  float4 o;
  o.x = (v.x - mu) * rstd * gg.x + bb.x;
  o.y = (v.y - mu) * rstd * gg.y + bb.y;
  o.z = (v.z - mu) * rstd * gg.z + bb.z;
  o.w = (v.w - mu) * rstd * gg.w + bb.w;
  *(float4*)(out + base) = o;
}

// ---------------------------------------------------------------------------
extern "C" void kernel_launch(void* const* d_in, const int* in_sizes, int n_in,
                              void* d_out, int out_size, void* d_ws, size_t ws_size,
                              hipStream_t stream) {
  (void)in_sizes; (void)n_in; (void)out_size; (void)ws_size;

  const float* x      = (const float*)d_in[0];
  const float* w_in   = (const float*)d_in[1];
  const float* b_in   = (const float*)d_in[2];
  const float* conv_w = (const float*)d_in[3];
  const float* conv_b = (const float*)d_in[4];
  // d_in[5]=w_xp, d_in[6]=b_xp, d_in[9]=A_log: dead in the reference
  const float* w_dt   = (const float*)d_in[7];
  const float* b_dt   = (const float*)d_in[8];
  const float* Dskip  = (const float*)d_in[10];
  const float* w_out  = (const float*)d_in[11];
  const float* b_out  = (const float*)d_in[12];
  const float* ln_g   = (const float*)d_in[13];
  const float* ln_b   = (const float*)d_in[14];
  float* out = (float*)d_out;

  const size_t EI = (size_t)M_ROWS * D_INNER;      // 33,554,432
  const size_t EM = (size_t)M_ROWS * D_MODEL;      // 16,777,216

  char* ws = (char*)d_ws;
  float*  u       = (float*)(ws);                      // [EI] f32 (dead after conv)
  float*  alpha   = u;                                 // alias
  float*  g       = (float*)(ws + EI * 4);             // [EI] f32
  __bf16* xs      = (__bf16*)(ws + EI * 8);            // [EI] bf16 (dead after scan)
  float*  h       = (float*)(ws + EI * 8);             // alias: [EM] f32 pre-LN
  __bf16* yy      = (__bf16*)(ws + EI * 8 + EI * 2);   // [EI] bf16
  __bf16* x_bf    = (__bf16*)(ws + EI * 8 + EI * 4);   // [EM] bf16
  char*   wbase   = ws + EI * 8 + EI * 4 + EM * 2;
  __bf16* w_in_t  = (__bf16*)(wbase);                                  // 4096x1024
  __bf16* w_dt_t  = (__bf16*)(wbase + (size_t)4096 * 1024 * 2);        // 2048x2048
  __bf16* w_out_t = (__bf16*)(wbase + (size_t)4096 * 1024 * 2
                                    + (size_t)2048 * 2048 * 2);        // 1024x2048

  const dim3 tb(32, 8);
  // 0) activations x -> bf16; weights -> bf16 transposed (N x K)
  cvt_bf16_kernel<<<(unsigned)(EM / 1024), 256, 0, stream>>>(x, x_bf);
  transpose_cvt_kernel<<<dim3(4096 / 32, 1024 / 32), tb, 0, stream>>>(w_in, w_in_t, 1024, 4096);
  transpose_cvt_kernel<<<dim3(2048 / 32, 2048 / 32), tb, 0, stream>>>(w_dt, w_dt_t, 2048, 2048);
  transpose_cvt_kernel<<<dim3(1024 / 32, 2048 / 32), tb, 0, stream>>>(w_out, w_out_t, 2048, 1024);

  // 1) in-proj GEMM + silu split -> u (f32), g (f32)
  gemm_wmma_kernel<0><<<dim3(4096 / BN, M_ROWS / BM), 256, 0, stream>>>(
      x_bf, w_in_t, b_in, u, g, nullptr, M_ROWS, 4096, D_MODEL);

  // 2) depthwise causal conv + silu -> xs (bf16)
  conv_silu_kernel<<<(unsigned)(EI / 256), 256, 0, stream>>>(u, conv_w, conv_b, xs);

  // 3) dt-proj GEMM + softplus + sigmoid -> alpha (aliases u)
  gemm_wmma_kernel<1><<<dim3(2048 / BN, M_ROWS / BM), 256, 0, stream>>>(
      xs, w_dt_t, b_dt, alpha, nullptr, nullptr, M_ROWS, D_INNER, D_INNER);

  // 4) gated EMA scan + skip + gate -> yy (bf16)
  scan_kernel<<<(BATCH * D_INNER) / 256, 256, 0, stream>>>(xs, alpha, g, Dskip, yy);

  // 5) out-proj GEMM + bias + residual -> h (aliases xs)
  gemm_wmma_kernel<2><<<dim3(1024 / BN, M_ROWS / BM), 256, 0, stream>>>(
      yy, w_out_t, b_out, h, nullptr, x, M_ROWS, D_MODEL, D_INNER);

  // 6) LayerNorm -> out
  layernorm_kernel<<<M_ROWS, 256, 0, stream>>>(h, ln_g, ln_b, out);
}
